// ARBlock_11269994184853
// MI455X (gfx1250) — matrix operand order repaired and checked by reading, not compiled, not run on hardware
//
#include <hip/hip_runtime.h>
#include <hip/hip_bf16.h>

typedef float v8f __attribute__((ext_vector_type(8)));
typedef unsigned int v8u __attribute__((ext_vector_type(8)));
typedef __bf16 v16bf __attribute__((ext_vector_type(16)));
typedef unsigned short ushort_t;
typedef unsigned int uint_t;

constexpr int Bb = 32;
constexpr int Ss = 2048;
constexpr int Dd = 512;
constexpr int F4 = 4 * Dd;          // 2048
constexpr int Mrows = Bb * Ss;      // 65536

__device__ __forceinline__ float sigmoidf(float x) {
    return 1.0f / (1.0f + __expf(-x));
}

__device__ __forceinline__ float gelu_tanh(float v) {
    float v3 = v * v * v;
    return 0.5f * v * (1.0f + tanhf(0.7978845608028654f * (v + 0.044715f * v3)));
}

// Split fp32 into hi/lo bf16 halves (RNE):  v ~= hi + lo
__device__ __forceinline__ void split_bf16(float v, ushort_t& h, ushort_t& l) {
    __hip_bfloat16 bh = __float2bfloat16(v);
    h = __builtin_bit_cast(ushort_t, bh);
    float rem = v - __bfloat162float(bh);
    l = __builtin_bit_cast(ushort_t, __float2bfloat16(rem));
}

// Low 32 bits of a flat shared pointer == wave-relative LDS byte address
__device__ __forceinline__ unsigned lds_addr(const void* p) {
    return (unsigned)(uintptr_t)p;
}

// ---------------------------------------------------------------------------
// Per-row LayerNorm statistics: stats[row] = {mu, rsqrt(var+eps)}
// ---------------------------------------------------------------------------
__global__ void ln_stats_kernel(const float* __restrict__ x, float* __restrict__ stats) {
    int wave = threadIdx.x >> 5;
    int lane = threadIdx.x & 31;
    int row = blockIdx.x * 8 + wave;
    const float* xr = x + (size_t)row * Dd;
    float s = 0.f, s2 = 0.f;
    #pragma unroll
    for (int i = 0; i < Dd / 32; ++i) {
        float v = xr[lane + 32 * i];
        s += v;
        s2 += v * v;
    }
    #pragma unroll
    for (int m = 16; m > 0; m >>= 1) {
        s  += __shfl_xor(s,  m, 32);
        s2 += __shfl_xor(s2, m, 32);
    }
    if (lane == 0) {
        float mu  = s * (1.0f / Dd);
        float var = s2 * (1.0f / Dd) - mu * mu;
        stats[2 * row]     = mu;
        stats[2 * row + 1] = rsqrtf(var + 1e-6f);
    }
}

// ---------------------------------------------------------------------------
// Weight pre-split+transpose: f32 [K][N] -> bf16 hi/lo packed [N][K].
// K-pairs for one output column become contiguous uint pairs (ISA B-frag
// layout) and B tile staging becomes a straight copy -> async-to-LDS capable.
// ---------------------------------------------------------------------------
__global__ void split_transpose_kernel(const float* __restrict__ in,
                                       ushort_t* __restrict__ hiT, ushort_t* __restrict__ loT,
                                       int K, int N) {
    int o = blockIdx.x * 256 + threadIdx.x;   // output index over [N][K]
    if (o < K * N) {
        int n = o / K, k = o - n * K;
        split_bf16(in[(size_t)k * N + n], hiT[o], loT[o]);
    }
}

// ---------------------------------------------------------------------------
// Tiled split-bf16 WMMA GEMM: C = op( A' @ B + bias ), A' = A or LN(A).
//   a*b ~= a_hi*b_hi + a_hi*b_lo + a_lo*b_hi  (3x v_wmma_f32_16x16x32_bf16)
// B supplied pre-split transposed-packed bf16 [N][K]; staged to LDS with
// double-buffered GLOBAL_LOAD_ASYNC_TO_LDS_B64 (ASYNCcnt), 1 barrier/slab.
//  AMODE: 0 = raw A, 1 = LayerNorm while staging
//  EPI  : 0 = store acc+bias, 1 = store gelu(acc+bias), 2 = C += acc+bias
// ---------------------------------------------------------------------------
template <int AMODE, int EPI>
__global__ __launch_bounds__(512)
void wmma_gemm_bf16x3_kernel(const float* __restrict__ A,
                             const ushort_t* __restrict__ BThi, const ushort_t* __restrict__ BTlo,
                             const float* __restrict__ bias, float* __restrict__ C,
                             const float* __restrict__ stats, const float* __restrict__ gamma,
                             const float* __restrict__ beta,
                             int M, int N, int K) {
    constexpr int BM = 64, BN = 64, BK = 32;
    constexpr int ASTR = 36;  // ushorts (18 uints): conflict-free A-frag reads
    constexpr int BSTR = 52;  // ushorts (26 uints, 8B-aligned rows): conflict-free B-frag reads
    __shared__ ushort_t AsHi[2][BM][ASTR], AsLo[2][BM][ASTR];
    __shared__ ushort_t BsHi[2][BN][BSTR], BsLo[2][BN][BSTR];  // [n][k] layout

    const int tid  = threadIdx.x;
    const int lane = tid & 31;
    const int w    = tid >> 5;     // 0..15
    const int wm   = w >> 2;
    const int wn   = w & 3;
    const int row0 = blockIdx.y * BM;
    const int col0 = blockIdx.x * BN;

    const int ln     = lane & 15;
    const int khalf4 = (lane >> 4) * 4;
    const int khalf8 = (lane >> 4) * 8;
    const int hi8    = (lane >> 4) << 3;

    v8f acc = {};
    v8f accL = {};

    // ---- async B staging setup: 64 rows x 8 qwords x {hi,lo}, 1 qword each ----
    const int br = tid >> 3;              // 0..63: n within tile
    const int bq = tid & 7;               // 8-byte chunk within 64B row-slab
    const ushort_t* gBh = BThi + (size_t)(col0 + br) * K + bq * 4;
    const ushort_t* gBl = BTlo + (size_t)(col0 + br) * K + bq * 4;
    const unsigned ldsBh0 = lds_addr(&BsHi[0][br][bq * 4]);
    const unsigned ldsBh1 = lds_addr(&BsHi[1][br][bq * 4]);
    const unsigned ldsBl0 = lds_addr(&BsLo[0][br][bq * 4]);
    const unsigned ldsBl1 = lds_addr(&BsLo[1][br][bq * 4]);

    auto issueB = [&](int k0, int buf) {
        const ushort_t* sH = gBh + k0;
        const ushort_t* sL = gBl + k0;
        unsigned dH = buf ? ldsBh1 : ldsBh0;
        unsigned dL = buf ? ldsBl1 : ldsBl0;
        asm volatile("global_load_async_to_lds_b64 %0, %1, off"
                     :: "v"(dH), "v"(sH) : "memory");
        asm volatile("global_load_async_to_lds_b64 %0, %1, off"
                     :: "v"(dL), "v"(sL) : "memory");
    };

    issueB(0, 0);   // prologue: B slab 0 in flight

    const int nch = K / BK;
    for (int i = 0; i < nch; ++i) {
        const int cur = i & 1;
        const int k0 = i * BK;

        // ---- stage A slab: LN transform + hi/lo split (64x32 f32, 4/thread) ----
        #pragma unroll
        for (int s = 0; s < 4; ++s) {
            int idx = tid + s * 512;
            int r = idx >> 5, c = idx & 31;
            float v = A[(size_t)(row0 + r) * K + (k0 + c)];
            if (AMODE == 1) {
                float mu = stats[2 * (row0 + r)];
                float rs = stats[2 * (row0 + r) + 1];
                v = (v - mu) * rs * gamma[k0 + c] + beta[k0 + c];
            }
            split_bf16(v, AsHi[cur][r][c], AsLo[cur][r][c]);
            if (k0 + BK < K)
                __builtin_prefetch(&A[(size_t)(row0 + r) * K + (k0 + BK + c)], 0, 0);
        }

        // my async copies for slab i have landed; then all waves rendezvous
        asm volatile("s_wait_asynccnt 0x0" ::: "memory");
        __syncthreads();

        // overwrite of buf cur^1 is safe only after barrier(i): all waves'
        // fragment reads of slab i-1 completed before they signaled it
        if (i + 1 < nch) issueB(k0 + BK, cur ^ 1);

        // ---- fragments (ISA 16-bit A/B layouts) + 3x WMMA ----
        const uint_t* arH = (const uint_t*)&AsHi[cur][wm * 16 + ln][0];
        const uint_t* arL = (const uint_t*)&AsLo[cur][wm * 16 + ln][0];
        const uint_t* brH = (const uint_t*)&BsHi[cur][wn * 16 + ln][0];
        const uint_t* brL = (const uint_t*)&BsLo[cur][wn * 16 + ln][0];
        v8u auh, aul, buh, bul;
        #pragma unroll
        for (int v = 0; v < 8; ++v) {
            int ai = (v & 3) + ((v >> 2) << 3) + khalf4;
            int bi = (v & 3) + ((v >> 2) << 3) + khalf4;
            auh[v] = arH[ai];
            aul[v] = arL[ai];
            buh[v] = brH[bi];
            bul[v] = brL[bi];
        }
        v16bf ah = __builtin_bit_cast(v16bf, auh);
        v16bf al = __builtin_bit_cast(v16bf, aul);
        v16bf bh = __builtin_bit_cast(v16bf, buh);
        v16bf bl = __builtin_bit_cast(v16bf, bul);

        acc  = __builtin_amdgcn_wmma_f32_16x16x32_bf16(false, ah, false, bh, (short)0, acc,  false, false);
        accL = __builtin_amdgcn_wmma_f32_16x16x32_bf16(false, ah, false, bl, (short)0, accL, false, false);
        accL = __builtin_amdgcn_wmma_f32_16x16x32_bf16(false, al, false, bh, (short)0, accL, false, false);
    }

    // ---- epilogue ----
    const int n = col0 + wn * 16 + ln;
    const float bn = bias[n];
    #pragma unroll
    for (int r = 0; r < 8; ++r) {
        int m = row0 + wm * 16 + r + hi8;
        size_t off = (size_t)m * N + n;
        float v = acc[r] + accL[r] + bn;
        if (EPI == 0)      C[off] = v;
        else if (EPI == 1) C[off] = gelu_tanh(v);
        else               C[off] = C[off] + v;
    }
}

// ---------------------------------------------------------------------------
// One LSTM timestep: gates = xg[:,t,:] + h @ Wh ; c,h update ; out = x + h.
// h kept as bf16 hi/lo ping-pong buffers; Wh pre-split transposed [N][K].
// Fragments come straight from global (L2-resident) as b128 loads.
// ---------------------------------------------------------------------------
__global__ __launch_bounds__(256)
void lstm_step_kernel(const float* __restrict__ xg,
                      const ushort_t* __restrict__ whT_hi, const ushort_t* __restrict__ whT_lo,
                      const float* __restrict__ xin, float* __restrict__ out,
                      const ushort_t* __restrict__ hin_hi, const ushort_t* __restrict__ hin_lo,
                      ushort_t* __restrict__ hout_hi, ushort_t* __restrict__ hout_lo,
                      float* __restrict__ cstate, int t) {
    __shared__ float gs[4][32][17];  // [gate][batch][col]

    const int tid  = threadIdx.x;
    const int lane = tid & 31;
    const int w    = tid >> 5;
    const int j    = w & 3;        // gate: i,f,g,o
    const int mt   = w >> 2;       // batch tile 0/1
    const int n0   = blockIdx.x * 16;

    const int ln     = lane & 15;
    const int khalf4 = (lane >> 4) * 4;
    const int khalf8 = (lane >> 4) * 8;
    const int hi8    = (lane >> 4) << 3;

    v8f acc;   // seeded with input projection (bias folded in)
    #pragma unroll
    for (int r = 0; r < 8; ++r) {
        int bb = mt * 16 + r + hi8;
        acc[r] = xg[((size_t)bb * Ss + t) * F4 + j * Dd + n0 + ln];
    }
    v8f accL = {};

    const uint_t* arH = (const uint_t*)hin_hi + (size_t)(mt * 16 + ln) * (Dd / 2);
    const uint_t* arL = (const uint_t*)hin_lo + (size_t)(mt * 16 + ln) * (Dd / 2);
    const int gcol = j * Dd + n0 + ln;
    const uint_t* brH = (const uint_t*)whT_hi + (size_t)gcol * (Dd / 2);
    const uint_t* brL = (const uint_t*)whT_lo + (size_t)gcol * (Dd / 2);

    for (int k0 = 0; k0 < Dd; k0 += 32) {
        const int base = k0 >> 1;
        v8u auh, aul, buh, bul;
        #pragma unroll
        for (int v = 0; v < 8; ++v) {
            int ai = base + (v & 3) + ((v >> 2) << 3) + khalf4;
            int bi = base + (v & 3) + ((v >> 2) << 3) + khalf4;
            auh[v] = arH[ai];
            aul[v] = arL[ai];
            buh[v] = brH[bi];
            bul[v] = brL[bi];
        }
        v16bf ah = __builtin_bit_cast(v16bf, auh);
        v16bf al = __builtin_bit_cast(v16bf, aul);
        v16bf bh = __builtin_bit_cast(v16bf, buh);
        v16bf bl = __builtin_bit_cast(v16bf, bul);
        acc  = __builtin_amdgcn_wmma_f32_16x16x32_bf16(false, ah, false, bh, (short)0, acc,  false, false);
        accL = __builtin_amdgcn_wmma_f32_16x16x32_bf16(false, ah, false, bl, (short)0, accL, false, false);
        accL = __builtin_amdgcn_wmma_f32_16x16x32_bf16(false, al, false, bh, (short)0, accL, false, false);
    }

    #pragma unroll
    for (int r = 0; r < 8; ++r)
        gs[j][mt * 16 + r + hi8][ln] = acc[r] + accL[r];
    __syncthreads();

    #pragma unroll
    for (int e = tid; e < Bb * 16; e += 256) {
        int bb = e >> 4, nn = e & 15;
        int col = n0 + nn;
        float ig = sigmoidf(gs[0][bb][nn]);
        float fg = sigmoidf(gs[1][bb][nn]);
        float gg = tanhf(gs[2][bb][nn]);
        float og = sigmoidf(gs[3][bb][nn]);
        size_t ci = (size_t)bb * Dd + col;
        float c = fg * cstate[ci] + ig * gg;
        cstate[ci] = c;
        float h = og * tanhf(c);
        split_bf16(h, hout_hi[ci], hout_lo[ci]);
        size_t xo = ((size_t)bb * Ss + t) * Dd + col;
        out[xo] = xin[xo] + h;
    }
}

__global__ void init_state_kernel(float* __restrict__ c,
                                  uint_t* __restrict__ h0hi, uint_t* __restrict__ h0lo) {
    int i = blockIdx.x * 256 + threadIdx.x;   // 16384 threads
    c[i] = 0.0f;
    if (i < (Bb * Dd) / 2) {
        h0hi[i] = 0u;
        h0lo[i] = 0u;
    }
}

// ---------------------------------------------------------------------------
extern "C" void kernel_launch(void* const* d_in, const int* in_sizes, int n_in,
                              void* d_out, int out_size, void* d_ws, size_t ws_size,
                              hipStream_t stream) {
    const float* x     = (const float*)d_in[0];
    const float* g1    = (const float*)d_in[1];
    const float* be1   = (const float*)d_in[2];
    const float* Wi    = (const float*)d_in[3];
    const float* Wh    = (const float*)d_in[4];
    const float* blstm = (const float*)d_in[5];
    const float* g2    = (const float*)d_in[6];
    const float* be2   = (const float*)d_in[7];
    const float* W1    = (const float*)d_in[8];
    const float* b1    = (const float*)d_in[9];
    const float* W2    = (const float*)d_in[10];
    const float* b2    = (const float*)d_in[11];
    float* out = (float*)d_out;

    // ---- workspace layout ----
    float* ws     = (float*)d_ws;
    float* xg     = ws;                                   // B*S*4D f32 (reused by FFN mid)
    float* stats1 = xg + (size_t)Mrows * F4;
    float* stats2 = stats1 + 2 * (size_t)Mrows;
    float* cst    = stats2 + 2 * (size_t)Mrows;           // B*D f32
    ushort_t* bws = (ushort_t*)(cst + (size_t)Bb * Dd);   // bf16 region
    const size_t HS = (size_t)Bb * Dd;                    // 16384
    const size_t WS = (size_t)Dd * F4;                    // 1048576
    ushort_t* h_hi[2] = { bws,      bws + 2 * HS };
    ushort_t* h_lo[2] = { bws + HS, bws + 3 * HS };
    ushort_t* wiT_hi = bws + 4 * HS;
    ushort_t* wiT_lo = wiT_hi + WS;
    ushort_t* whT_hi = wiT_lo + WS;
    ushort_t* whT_lo = whT_hi + WS;
    ushort_t* w1T_hi = whT_lo + WS;
    ushort_t* w1T_lo = w1T_hi + WS;
    ushort_t* w2T_hi = w1T_lo + WS;
    ushort_t* w2T_lo = w2T_hi + WS;

    // ---- phase 0: pre-split all weights into transposed-packed bf16 hi/lo ----
    const int nW = (int)WS;
    split_transpose_kernel<<<nW / 256, 256, 0, stream>>>(Wi, wiT_hi, wiT_lo, Dd, F4);
    split_transpose_kernel<<<nW / 256, 256, 0, stream>>>(Wh, whT_hi, whT_lo, Dd, F4);
    split_transpose_kernel<<<nW / 256, 256, 0, stream>>>(W1, w1T_hi, w1T_lo, Dd, F4);
    split_transpose_kernel<<<nW / 256, 256, 0, stream>>>(W2, w2T_hi, w2T_lo, F4, Dd);

    // ---- phase 1: LN1 stats, xg = LN1(x) @ Wi + b_lstm ----
    ln_stats_kernel<<<Mrows / 8, 256, 0, stream>>>(x, stats1);
    wmma_gemm_bf16x3_kernel<1, 0><<<dim3(F4 / 64, Mrows / 64), 512, 0, stream>>>(
        x, wiT_hi, wiT_lo, blstm, xg, stats1, g1, be1, Mrows, F4, Dd);

    // ---- phase 2: sequential LSTM scan ----
    init_state_kernel<<<(Bb * Dd) / 256, 256, 0, stream>>>(cst, (uint_t*)h_hi[0], (uint_t*)h_lo[0]);
    for (int t = 0; t < Ss; ++t) {
        int pi = t & 1, po = pi ^ 1;
        lstm_step_kernel<<<Dd / 16, 256, 0, stream>>>(
            xg, whT_hi, whT_lo, x, out,
            h_hi[pi], h_lo[pi], h_hi[po], h_lo[po], cst, t);
    }
    // out now holds x1 = x + lstm(...)

    // ---- phase 3: FFN ----
    ln_stats_kernel<<<Mrows / 8, 256, 0, stream>>>(out, stats2);
    wmma_gemm_bf16x3_kernel<1, 1><<<dim3(F4 / 64, Mrows / 64), 512, 0, stream>>>(
        out, w1T_hi, w1T_lo, b1, xg, stats2, g2, be2, Mrows, F4, Dd);
    wmma_gemm_bf16x3_kernel<0, 2><<<dim3(Dd / 64, Mrows / 64), 512, 0, stream>>>(
        xg, w2T_hi, w2T_lo, b2, out, nullptr, nullptr, nullptr, Mrows, Dd, F4);
}
